// MHAttn_88218628260152
// MI455X (gfx1250) — matrix operand order repaired and checked by reading, not compiled, not run on hardware
//
#include <hip/hip_runtime.h>

#define BATCH   4
#define SEQ     2048
#define DIMS    512
#define NHEADS  8
#define KDIM    64
#define VDIM    512

typedef __attribute__((ext_vector_type(16))) __bf16 v16bf;
typedef __attribute__((ext_vector_type(8)))  float  v8f;

union V16 { v16bf v; uint4 q[2]; unsigned short u[16]; };

__device__ __forceinline__ unsigned short f2bf(float f) {
  unsigned int x = __float_as_uint(f);
  return (unsigned short)((x + 0x7FFFu + ((x >> 16) & 1u)) >> 16);
}

__device__ __forceinline__ v8f vzero8() {
  v8f z = {0.f, 0.f, 0.f, 0.f, 0.f, 0.f, 0.f, 0.f};
  return z;
}

__device__ __forceinline__ v8f wmma_bf16(v16bf a, v16bf b, v8f c) {
  // D = A(16x32 bf16) * B(32x16 bf16) + C(16x16 f32)
  return __builtin_amdgcn_wmma_f32_16x16x32_bf16(false, a, false, b, (short)0, c,
                                                 false, false);
}

// ---- WMMA operand loaders (wave32 VGPR layouts, CDNA5 ISA 7.12.2) -----------
// All loads below are 16-byte vector loads from bf16 (ushort) storage.

// A tile (16 rows x 32 k), row-major source. lane&15 = row;
// lanes 0-15 take k {0..7,16..23}, lanes 16-31 take {8..15,24..31}.
__device__ __forceinline__ v16bf load_a_bf16(const unsigned short* __restrict__ base,
                                             int lda, int kc) {
  const int lane = threadIdx.x & 31, hl = lane >> 4, r = lane & 15;
  const unsigned short* p = base + (size_t)r * lda + kc + hl * 8;
  V16 t;
  t.q[0] = *(const uint4*)(p);
  t.q[1] = *(const uint4*)(p + 16);
  return t.v;
}

// B tile (32 k x 16 n) where each column n is contiguous in k:
// element (k, n) at src[(n0+n)*ld + k0 + k]. lane&15 = col;
// lanes 0-15 hold k {0..7,16..23}, lanes 16-31 hold k {8..15,24..31}.
__device__ __forceinline__ v16bf load_b_bf16(const unsigned short* __restrict__ src,
                                             int ld, int k0, int n0) {
  const int lane = threadIdx.x & 31, hl = lane >> 4, n = lane & 15;
  const unsigned short* p = src + (size_t)(n0 + n) * ld + k0 + hl * 8;
  V16 t;
  t.q[0] = *(const uint4*)(p);
  t.q[1] = *(const uint4*)(p + 16);
  return t.v;
}

// ---- conversion kernels (run once; removes bf16 packing from GEMM loops) ----

// fp32 -> bf16, vectorized: float4 in, 4 packed bf16 (uint2) out.
__global__ __launch_bounds__(256) void cvt_bf16_kernel(const float* __restrict__ in,
                                                       unsigned short* __restrict__ out,
                                                       int n4) {
  const int i = blockIdx.x * 256 + threadIdx.x;
  if (i >= n4) return;
  const float4 f = ((const float4*)in)[i];
  uint2 r;
  r.x = (unsigned)f2bf(f.x) | ((unsigned)f2bf(f.y) << 16);
  r.y = (unsigned)f2bf(f.z) | ((unsigned)f2bf(f.w) << 16);
  ((uint2*)out)[i] = r;
}

// W[k][n] fp32 (row-major, N=512) -> Wt[n][k] bf16 (k contiguous per column).
__global__ __launch_bounds__(256) void cvt_wt_kernel(const float* __restrict__ W,
                                                     unsigned short* __restrict__ Wt,
                                                     int K) {
  const int idx = blockIdx.x * 256 + threadIdx.x;   // coalesced read of W
  if (idx >= K * DIMS) return;
  const int k = idx >> 9, n = idx & (DIMS - 1);
  Wt[(size_t)n * K + k] = f2bf(W[idx]);
}

// ---- kernel 1: input projections (Q/K de-interleaved, V transposed) ---------
// A: bf16 activations [8192][512]; B: Wt bf16 [512 n][512 k].
// Each wave computes a 16x64 output strip (4 accum tiles, A reused 4x).

__global__ __launch_bounds__(128) void proj_gemm_kernel(
    const unsigned short* __restrict__ Abf, const unsigned short* __restrict__ Wt,
    const float* __restrict__ bias, unsigned short* __restrict__ out, int mode) {
  const int wave = threadIdx.x >> 5, lane = threadIdx.x & 31;
  const int hl = lane >> 4, n = lane & 15;
  const int tile = blockIdx.x * 4 + wave;          // 512 row-tiles x 8 col-groups
  const int tm = tile >> 3, tg = tile & 7;
  const int row0 = tm * 16, n0g = tg * 64;
  const unsigned short* Abase = Abf + (size_t)row0 * DIMS;

  v8f acc[4];
#pragma unroll
  for (int j = 0; j < 4; ++j) acc[j] = vzero8();

  for (int kc = 0; kc < DIMS; kc += 32) {
    const v16bf a = load_a_bf16(Abase, DIMS, kc);
#pragma unroll
    for (int j = 0; j < 4; ++j)
      acc[j] = wmma_bf16(a, load_b_bf16(Wt, DIMS, kc, n0g + j * 16), acc[j]);
  }

#pragma unroll
  for (int j = 0; j < 4; ++j) {
    const int col = n0g + j * 16 + n;
    const float bb = bias[col];
#pragma unroll
    for (int v = 0; v < 8; ++v) {
      const int row = row0 + v + hl * 8;           // C layout: M = vgpr + 8*half
      const int b = row >> 11, l = row & (SEQ - 1);
      const float val = acc[j][v] + bb;
      size_t idx;
      if (mode == 0) {                             // Q/K: col -> (h=col&7, kd=col>>3)
        const int h = col & 7, kd = col >> 3;
        idx = (((size_t)b * NHEADS + h) * SEQ + l) * KDIM + kd;
      } else {                                     // V transposed: [b][vd][m]
        idx = ((size_t)b * VDIM + col) * SEQ + l;
      }
      out[idx] = f2bf(val);
    }
  }
}

// ---- kernel 2: causal flash attention, one block per (b, h, 16 rows) --------

__global__ __launch_bounds__(128) void attn_kernel(
    const unsigned short* __restrict__ Qh, const unsigned short* __restrict__ Kh,
    const unsigned short* __restrict__ Vt, unsigned short* __restrict__ Obf) {
  __shared__ __attribute__((aligned(32))) unsigned short P_lds[16][32];
  __shared__ float fac_lds[16];
  __shared__ float sum_lds[16];

  const int wave = threadIdx.x >> 5, lane = threadIdx.x & 31;
  const int hl = lane >> 4, n = lane & 15;
  const int lt = blockIdx.x & 127;
  const int h  = (blockIdx.x >> 7) & 7;
  const int b  = blockIdx.x >> 10;
  const int l0 = lt * 16;

  const unsigned short* Qbase = Qh + (((size_t)b * NHEADS + h) * SEQ + l0) * KDIM;
  const unsigned short* Krows = Kh + (((size_t)b * NHEADS + h) * SEQ) * KDIM;
  const unsigned short* Vbase = Vt + (size_t)b * VDIM * SEQ;

  const v16bf qa0 = load_a_bf16(Qbase, KDIM, 0);
  const v16bf qa1 = load_a_bf16(Qbase, KDIM, 32);

  v8f acc[8];
#pragma unroll
  for (int t = 0; t < 8; ++t) acc[t] = vzero8();
  float runM[8], runS[8];
#pragma unroll
  for (int v = 0; v < 8; ++v) { runM[v] = -1e30f; runS[v] = 0.f; }

  const int vd0 = wave * 128;                      // each wave: 128 of 512 V cols
  const int nb = (l0 + 16 + 31) >> 5;              // 32-key blocks to causal edge

  for (int kb = 0; kb < nb; ++kb) {
    const int key0 = kb * 32;
    __syncthreads();                               // previous PV consumed P_lds
    if (wave == 1 && lane == 0 && key0 + 32 < SEQ)
      __builtin_prefetch(Krows + (size_t)(key0 + 32) * KDIM, 0, 0);

    if (wave == 0) {
      const unsigned short* Kb = Krows + (size_t)key0 * KDIM;  // row=key, 64 kd
      v8f s0 = vzero8(), s1 = vzero8();
      s0 = wmma_bf16(qa0, load_b_bf16(Kb, KDIM, 0,  0),  s0);
      s0 = wmma_bf16(qa1, load_b_bf16(Kb, KDIM, 32, 0),  s0);
      s1 = wmma_bf16(qa0, load_b_bf16(Kb, KDIM, 0,  16), s1);
      s1 = wmma_bf16(qa1, load_b_bf16(Kb, KDIM, 32, 16), s1);
#pragma unroll
      for (int v = 0; v < 8; ++v) {
        const int rowg = l0 + v + hl * 8;
        const int c0 = key0 + n, c1 = c0 + 16;
        float ls0 = (c0 > rowg) ? -1e30f : s0[v] * 0.125f;   // 1/sqrt(64)
        float ls1 = (c1 > rowg) ? -1e30f : s1[v] * 0.125f;
        float mx = fmaxf(ls0, ls1);
        mx = fmaxf(mx, __shfl_xor(mx, 1));
        mx = fmaxf(mx, __shfl_xor(mx, 2));
        mx = fmaxf(mx, __shfl_xor(mx, 4));
        mx = fmaxf(mx, __shfl_xor(mx, 8));                   // stays in 16-lane half
        const float newM = fmaxf(runM[v], mx);
        const float f  = __expf(runM[v] - newM);
        const float p0 = __expf(ls0 - newM);
        const float p1 = __expf(ls1 - newM);
        float ts = p0 + p1;
        ts += __shfl_xor(ts, 1); ts += __shfl_xor(ts, 2);
        ts += __shfl_xor(ts, 4); ts += __shfl_xor(ts, 8);
        runS[v] = runS[v] * f + ts;
        runM[v] = newM;
        P_lds[v + hl * 8][n]      = f2bf(p0);
        P_lds[v + hl * 8][n + 16] = f2bf(p1);
        if (n == 0) fac_lds[v + hl * 8] = f;
      }
    }
    __syncthreads();

    const v16bf pa = load_a_bf16(&P_lds[0][0], 32, 0);       // P: 16 rows x 32 keys
    float myf[8];
#pragma unroll
    for (int v = 0; v < 8; ++v) myf[v] = fac_lds[v + hl * 8];
#pragma unroll
    for (int t = 0; t < 8; ++t) {
      v8f c = acc[t];
#pragma unroll
      for (int v = 0; v < 8; ++v) c[v] *= myf[v];            // flash rescale
      acc[t] = wmma_bf16(pa, load_b_bf16(Vbase + key0, SEQ, 0, vd0 + t * 16), c);
    }
  }

  if (wave == 0 && n == 0) {
#pragma unroll
    for (int v = 0; v < 8; ++v) sum_lds[v + hl * 8] = runS[v];
  }
  __syncthreads();
  float inv[8];
#pragma unroll
  for (int v = 0; v < 8; ++v) inv[v] = 1.f / sum_lds[v + hl * 8];
#pragma unroll
  for (int t = 0; t < 8; ++t) {
#pragma unroll
    for (int v = 0; v < 8; ++v) {
      const int row = l0 + v + hl * 8;
      const int col = vd0 + t * 16 + n;
      const size_t idx = ((size_t)b * SEQ + row) * (NHEADS * VDIM) + (size_t)h * VDIM + col;
      Obf[idx] = f2bf(acc[t][v] * inv[v]);
    }
  }
}

// ---- kernel 3: output projection [8192,4096] @ Wot(bf16) + bo ---------------

__global__ __launch_bounds__(128) void out_proj_kernel(
    const unsigned short* __restrict__ Obf, const unsigned short* __restrict__ Wot,
    const float* __restrict__ bo, float* __restrict__ out) {
  const int wave = threadIdx.x >> 5, lane = threadIdx.x & 31;
  const int hl = lane >> 4, n = lane & 15;
  const int tile = blockIdx.x * 4 + wave;          // 512 row-tiles x 8 col-groups
  const int tm = tile >> 3, tg = tile & 7;
  const int row0 = tm * 16, n0g = tg * 64;
  const int K = NHEADS * VDIM;                     // 4096
  const unsigned short* Abase = Obf + (size_t)row0 * K;

  v8f acc[4];
#pragma unroll
  for (int j = 0; j < 4; ++j) acc[j] = vzero8();

  for (int kc = 0; kc < K; kc += 32) {
    const v16bf a = load_a_bf16(Abase, K, kc);
#pragma unroll
    for (int j = 0; j < 4; ++j)
      acc[j] = wmma_bf16(a, load_b_bf16(Wot, K, kc, n0g + j * 16), acc[j]);
  }

#pragma unroll
  for (int j = 0; j < 4; ++j) {
    const int col = n0g + j * 16 + n;
    const float bb = bo[col];
#pragma unroll
    for (int v = 0; v < 8; ++v)
      out[(size_t)(row0 + v + hl * 8) * DIMS + col] = acc[j][v] + bb;
  }
}

// ---- host launcher ----------------------------------------------------------

extern "C" void kernel_launch(void* const* d_in, const int* in_sizes, int n_in,
                              void* d_out, int out_size, void* d_ws, size_t ws_size,
                              hipStream_t stream) {
  const float* query = (const float*)d_in[0];
  const float* key_  = (const float*)d_in[1];
  const float* value = (const float*)d_in[2];
  const float* Wq = (const float*)d_in[3];
  const float* bq = (const float*)d_in[4];
  const float* Wk = (const float*)d_in[5];
  const float* bk = (const float*)d_in[6];
  const float* Wv = (const float*)d_in[7];
  const float* bv = (const float*)d_in[8];
  const float* Wo = (const float*)d_in[9];
  const float* bo = (const float*)d_in[10];

  // Workspace layout (ushort units). Region A is used first for the bf16
  // activation staging (qbf|kbf|vbf, 12M) and later reused (stream-ordered)
  // for the 33.5M-element attention output Obf.
  const size_t ACT = (size_t)BATCH * SEQ * DIMS;            // 4M per tensor
  unsigned short* regionA = (unsigned short*)d_ws;
  unsigned short* qbf = regionA;
  unsigned short* kbf = regionA + ACT;
  unsigned short* vbf = regionA + 2 * ACT;
  unsigned short* Obf = regionA;                            // aliases qbf/kbf/vbf
  unsigned short* Qh  = regionA + (size_t)BATCH * SEQ * NHEADS * VDIM;  // after Obf
  unsigned short* Kh  = Qh + (size_t)BATCH * NHEADS * SEQ * KDIM;
  unsigned short* Vt  = Kh + (size_t)BATCH * NHEADS * SEQ * KDIM;
  unsigned short* Wqt = Vt + (size_t)BATCH * VDIM * SEQ;
  unsigned short* Wkt = Wqt + (size_t)DIMS * DIMS;
  unsigned short* Wvt = Wkt + (size_t)DIMS * DIMS;
  unsigned short* Wot = Wvt + (size_t)DIMS * DIMS;

  const dim3 blk256(256), blk128(128);
  const int n4 = (int)(ACT / 4);

  // one-shot fp32 -> bf16 conversions
  cvt_bf16_kernel<<<(n4 + 255) / 256, blk256, 0, stream>>>(query, qbf, n4);
  cvt_bf16_kernel<<<(n4 + 255) / 256, blk256, 0, stream>>>(key_,  kbf, n4);
  cvt_bf16_kernel<<<(n4 + 255) / 256, blk256, 0, stream>>>(value, vbf, n4);
  cvt_wt_kernel<<<(DIMS * DIMS + 255) / 256, blk256, 0, stream>>>(Wq, Wqt, DIMS);
  cvt_wt_kernel<<<(DIMS * DIMS + 255) / 256, blk256, 0, stream>>>(Wk, Wkt, DIMS);
  cvt_wt_kernel<<<(DIMS * DIMS + 255) / 256, blk256, 0, stream>>>(Wv, Wvt, DIMS);
  cvt_wt_kernel<<<(NHEADS * VDIM * DIMS + 255) / 256, blk256, 0, stream>>>(Wo, Wot,
                                                                NHEADS * VDIM);

  // projections (all-bf16 WMMA GEMMs)
  const int proj_blocks = (BATCH * SEQ / 16) * (DIMS / 64) / 4;   // 1024
  proj_gemm_kernel<<<proj_blocks, blk128, 0, stream>>>(qbf, Wqt, bq, Qh, 0);
  proj_gemm_kernel<<<proj_blocks, blk128, 0, stream>>>(kbf, Wkt, bk, Kh, 0);
  proj_gemm_kernel<<<proj_blocks, blk128, 0, stream>>>(vbf, Wvt, bv, Vt, 1);

  // attention (writes Obf over the dead activation staging region)
  attn_kernel<<<BATCH * NHEADS * (SEQ / 16), blk128, 0, stream>>>(Qh, Kh, Vt, Obf);

  // output projection
  out_proj_kernel<<<proj_blocks, blk128, 0, stream>>>(Obf, Wot, bo, (float*)d_out);

  (void)in_sizes; (void)n_in; (void)out_size; (void)ws_size;
}
